// PairmixBlock_79551384256576
// MI455X (gfx1250) — matrix operand order repaired and checked by reading, not compiled, not run on hardware
//
#include <hip/hip_runtime.h>
#include <hip/hip_bf16.h>

typedef __attribute__((ext_vector_type(16))) _Float16 v16h;
typedef __attribute__((ext_vector_type(8)))  _Float16 v8h;
typedef __attribute__((ext_vector_type(8)))  float    v8f;

#define N_NODES   10000
#define N_EDGES   80000
#define NTILE     160                     // 5 paths * 8 u * 4 v-pairs : 16-wide column tiles of w4
#define W4F_ELEMS (NTILE * 2 * 32 * 16)   // 163840 f16 : [tile][kchunk][lane][16 slots]
#define WBF_ELEMS (20 * 32 * 16)          // 10240  f16 : w1(4 frags) w2(8) w3(8)
#define CHUNK_T   8                       // tiles staged per async chunk
#define CHUNK_ELE (CHUNK_T * 2 * 32 * 16) // 8192 f16 = 16 KB per chunk
#define N_CHUNK   (NTILE / CHUNK_T)       // 20
#define SILU_C    1.679f
#define INV_128   0.08838834764831845f    // 1/sqrt(128)
#define RSQRT3    0.5773502691896258f
#define RSQRT8    0.35355339059327373f

// ---------------------------------------------------------------------------
// Pre-swizzle w_mlp4 (64 x 2560, row major) into WMMA 16-bit A-fragment layout
// (ISA 7.12.2: lane -> M = lane%16 ; lane half + vgpr -> K).  Scales folded in:
// 1/sqrt(64) (mlp), INV=1/sqrt(128), and 1/sqrt(3) for path 1.
// ---------------------------------------------------------------------------
__global__ void prep_w4_frags(const float* __restrict__ w4, _Float16* __restrict__ dst) {
  int gid  = blockIdx.x * 256 + threadIdx.x;   // [t][kc][lane][slot]
  int s    = gid & 15;
  int lane = (gid >> 4) & 31;
  int kc   = (gid >> 9) & 1;
  int t    = gid >> 10;
  int r    = s >> 1;                                   // VGPR index 0..7
  int kl   = ((lane & 16) ? 16 : 0) + s;
  int K    = kc * 32 + kl;                             // hidden index 0..63
  int col  = t * 16 + (lane & 15);                     // flat weight col 0..2559
  int p    = col >> 9;                                 // path = col/512
  float scale = 0.125f * INV_128 * (p == 1 ? RSQRT3 : 1.0f);
  dst[gid] = (_Float16)(w4[K * 2560 + col] * scale);
}

// ---------------------------------------------------------------------------
// Pre-swizzle w1 (8x64, zero-padded to K=32), w2, w3 (64x64) into WMMA 16-bit
// B-fragment layout (lane -> N = lane%16 ; lane half + slot -> K).
// frag ids: w1: 0..3 (nt, kc=0) ; w2: 4..11 (nt*2+kc) ; w3: 12..19
// ---------------------------------------------------------------------------
__global__ void prep_w123_frags(const float* __restrict__ w1, const float* __restrict__ w2,
                                const float* __restrict__ w3, _Float16* __restrict__ dst) {
  int gid  = blockIdx.x * 256 + threadIdx.x;   // [frag][lane][slot]
  int s    = gid & 15;
  int lane = (gid >> 4) & 31;
  int f    = gid >> 9;
  float val;
  if (f < 4) {
    int N = f * 16 + (lane & 15);
    int K = ((lane & 16) ? 16 : 0) + s;
    val   = (K < 8) ? w1[K * 64 + N] * RSQRT8 : 0.0f;
  } else {
    int ff = f - 4;
    const float* w = (ff < 8) ? w2 : w3;
    ff &= 7;
    int N = (ff >> 1) * 16 + (lane & 15);
    int K = (ff & 1) * 32 + ((s & 8) ? 16 : 0) + ((lane & 16) ? 8 : 0) + (s & 7);
    val   = w[K * 64 + N] * 0.125f;              // 1/sqrt(64)
  }
  dst[gid] = (_Float16)val;
}

__device__ __forceinline__ v8f wmma_f16(v16h a, v16h b, v8f c) {
  return __builtin_amdgcn_wmma_f32_16x16x32_f16(false, a, false, b, (short)0, c, false, false);
}
__device__ __forceinline__ float silu_s(float x) {
  return SILU_C * x / (1.0f + __expf(-x));
}

// --- CDNA5 async global->LDS copy (ASYNCcnt-tracked DMA, ISA 10.x / 15.18.3).
// vdst VGPR carries the LDS byte address (low 32 bits of the generic pointer
// per the LDS-aperture decode); vaddr is the 64-bit global address.
__device__ __forceinline__ void async_copy_b128(void* lds_dst, const void* gsrc) {
  unsigned loff = (unsigned)(unsigned long long)lds_dst;
  asm volatile("global_load_async_to_lds_b128 %0, %1, off"
               :: "v"(loff), "v"(gsrc) : "memory");
}
__device__ __forceinline__ void wait_async0() {
  asm volatile("s_wait_asynccnt 0x0" ::: "memory");
}
__device__ __forceinline__ void wait_async8() {   // first staged chunk arrived
  asm volatile("s_wait_asynccnt 0x8" ::: "memory");
}

// all 128 threads cooperatively issue one 16 KB chunk (8 async b128 per thread)
__device__ __forceinline__ void stage_chunk(const _Float16* __restrict__ w4f, int chunk,
                                            _Float16* dstbuf, int tid) {
  const _Float16* src = w4f + (size_t)chunk * CHUNK_ELE;
#pragma unroll
  for (int j = 0; j < 8; ++j) {
    int idx = (j * 128 + tid) * 8;               // 8 f16 = 16 B per op
    async_copy_b128(dstbuf + idx, src + idx);
  }
}

// ---------------------------------------------------------------------------
// Fused kernel: 4 waves / block, each wave owns 16 edges.
//   MLP (WMMA) -> h in LDS -> stream 160 w4 column tiles:
//   D[wcol, edge] = w4^T x h  (2x v_wmma_f32_16x16x32_f16), consumed in-register
//   into the tensor-product accumulators (lane = edge; rows r = output w).
//   w4 fragments are double-buffered block-wide in LDS via async DMA.
// ---------------------------------------------------------------------------
__global__ void __launch_bounds__(128) fctp_fused(
    const float* __restrict__ nodes, const float* __restrict__ efeat,
    const int* __restrict__ eidx, const _Float16* __restrict__ w4f,
    const _Float16* __restrict__ wbf, float* __restrict__ out) {
  __shared__ _Float16 sH[4][16 * 64];     // per-wave activations (edge, hid) f16
  __shared__ float    sF[4][16 * 68];     // per-wave node features, stride 68 (bank safe)
  __shared__ _Float16 sW[2][CHUNK_ELE];   // double-buffered w4 fragments (2 x 16 KB)

  const int tid   = threadIdx.x;
  const int lane  = tid & 31;
  const int wid   = tid >> 5;
  const int col16 = lane & 15;            // edge-in-group owned by this lane
  const int half  = lane >> 4;
  const int e     = (blockIdx.x * 4 + wid) * 16 + col16;

  // kick off the first two fragment chunks; the DMA overlaps the whole MLP phase
  stage_chunk(w4f, 0, sW[0], tid);
  stage_chunk(w4f, 1, sW[1], tid);

  // ---- stage node features: half 0 loads x1=receiver, half 1 loads x2=sender
  {
    int n = eidx[(half ? 0 : 1) * N_EDGES + e];      // row0=sender, row1=receiver
    const float* xp = nodes + n * 32;
    const int soff = half ? 8 : 0, voff = half ? 40 : 16;
#pragma unroll
    for (int j = 0; j < 8; ++j) {
      float4 q = *(const float4*)(xp + 4 * j);
      int off = (j < 2) ? (soff + 4 * j) : (voff + 4 * (j - 2));
      *(float4*)(&sF[wid][col16 * 68 + off]) = q;
    }
  }

  // ---- edge MLP, layer 1 (K padded 8->32) ----
  v8f cz = {};
  v16h a1;
#pragma unroll
  for (int s = 0; s < 16; ++s) a1[s] = (_Float16)0.f;
  if (half == 0) {                        // lanes 0-15 hold K=0..7 (real features)
    float4 f0 = *(const float4*)(efeat + e * 8);
    float4 f1 = *(const float4*)(efeat + e * 8 + 4);
    a1[0] = (_Float16)f0.x; a1[1] = (_Float16)f0.y; a1[2] = (_Float16)f0.z; a1[3] = (_Float16)f0.w;
    a1[4] = (_Float16)f1.x; a1[5] = (_Float16)f1.y; a1[6] = (_Float16)f1.z; a1[7] = (_Float16)f1.w;
  }
#pragma unroll
  for (int nt = 0; nt < 4; ++nt) {
    v16h b = *(const v16h*)(wbf + (nt * 32 + lane) * 16);
    v8f d = wmma_f16(a1, b, cz);
#pragma unroll
    for (int r = 0; r < 8; ++r)
      sH[wid][(half * 8 + r) * 64 + nt * 16 + col16] = (_Float16)silu_s(d[r]);
  }

  // ---- layers 2 and 3 (16x64 @ 64x64), LDS transpose between layers ----
#pragma unroll
  for (int L = 0; L < 2; ++L) {
    const int fb = 4 + L * 8;
    const _Float16* hp = &sH[wid][col16 * 64];
    const int ab = half ? 8 : 0;          // A layout: lane half splits K
    v8h p0 = *(const v8h*)(hp + ab);
    v8h p1 = *(const v8h*)(hp + ab + 16);
    v8h p2 = *(const v8h*)(hp + ab + 32);
    v8h p3 = *(const v8h*)(hp + ab + 48);
    v16h A0 = __builtin_shufflevector(p0, p1, 0,1,2,3,4,5,6,7,8,9,10,11,12,13,14,15);
    v16h A1 = __builtin_shufflevector(p2, p3, 0,1,2,3,4,5,6,7,8,9,10,11,12,13,14,15);
#pragma unroll
    for (int nt = 0; nt < 4; ++nt) {
      v16h b0 = *(const v16h*)(wbf + ((fb + nt * 2    ) * 32 + lane) * 16);
      v16h b1 = *(const v16h*)(wbf + ((fb + nt * 2 + 1) * 32 + lane) * 16);
      v8f d = wmma_f16(A0, b0, cz);
      d = wmma_f16(A1, b1, d);
#pragma unroll
      for (int r = 0; r < 8; ++r)
        sH[wid][(half * 8 + r) * 64 + nt * 16 + col16] = (_Float16)silu_s(d[r]);
    }
  }

  // ---- B fragments of h (K=hid, N=edge) for the big GEMM ----
  const _Float16* hb = &sH[wid][col16 * 64];
  const int kb = half ? 16 : 0;
  v16h B0 = *(const v16h*)(hb + kb);         // K  0..31 half
  v16h B1 = *(const v16h*)(hb + kb + 32);    // K 32..63 half

  // ---- fused tp-weight GEMM + tensor-product contraction ----
  float acc0[8], a1x[8], a1y[8], a1z[8], a2x[8], a2y[8], a2z[8];
#pragma unroll
  for (int r = 0; r < 8; ++r) {
    acc0[r] = 0.f; a1x[r] = a1y[r] = a1z[r] = 0.f; a2x[r] = a2y[r] = a2z[r] = 0.f;
  }
  const float* fp = &sF[wid][col16 * 68];    // [0..7]=s1 [8..15]=s2 [16..39]=v1 [40..63]=v2

  wait_async8();                // chunk 0 resident (chunk 1 still in flight)
  __syncthreads();

  for (int c = 0; c < N_CHUNK; ++c) {
    const _Float16* wbuf = sW[c & 1];
    for (int tt = 0; tt < CHUNK_T; ++tt) {
      const int t = c * CHUNK_T + tt;
      v16h A0 = *(const v16h*)(wbuf + ((tt * 2    ) * 32 + lane) * 16);
      v16h A1 = *(const v16h*)(wbuf + ((tt * 2 + 1) * 32 + lane) * 16);
      v8f d = wmma_f16(A0, B0, cz);
      d = wmma_f16(A1, B1, d);
      // lane holds tp_weights[e, t*16 + r + 8*half] => (p, u, v = 2*vp+half, w=r)
      const int p = t >> 5;
      const int u = (t >> 2) & 7;
      const int v = ((t & 3) << 1) + half;
      if (p == 0) {
        float pv = fp[u] * fp[8 + v];
#pragma unroll
        for (int r = 0; r < 8; ++r) acc0[r] += d[r] * pv;
      } else if (p == 1) {                 // 1/sqrt(3) folded into w4 scale
        float pv = fp[16 + 3*u] * fp[40 + 3*v] + fp[17 + 3*u] * fp[41 + 3*v]
                 + fp[18 + 3*u] * fp[42 + 3*v];
#pragma unroll
        for (int r = 0; r < 8; ++r) acc0[r] += d[r] * pv;
      } else if (p == 2) {
        float s1u = fp[u];
        float cx = s1u * fp[40 + 3*v], cy = s1u * fp[41 + 3*v], cc = s1u * fp[42 + 3*v];
#pragma unroll
        for (int r = 0; r < 8; ++r) { a1x[r] += d[r]*cx; a1y[r] += d[r]*cy; a1z[r] += d[r]*cc; }
      } else if (p == 3) {
        float s2v = fp[8 + v];
        float cx = fp[16 + 3*u] * s2v, cy = fp[17 + 3*u] * s2v, cc = fp[18 + 3*u] * s2v;
#pragma unroll
        for (int r = 0; r < 8; ++r) { a1x[r] += d[r]*cx; a1y[r] += d[r]*cy; a1z[r] += d[r]*cc; }
      } else {                             // cross(v1[u], v2[v])
        float ax = fp[16 + 3*u], ay = fp[17 + 3*u], az = fp[18 + 3*u];
        float bx = fp[40 + 3*v], by = fp[41 + 3*v], bz = fp[42 + 3*v];
        float cx = ay * bz - az * by;
        float cy = az * bx - ax * bz;
        float cc = ax * by - ay * bx;
#pragma unroll
        for (int r = 0; r < 8; ++r) { a2x[r] += d[r]*cx; a2y[r] += d[r]*cy; a2z[r] += d[r]*cc; }
      }
    }
    if (c < N_CHUNK - 1) {
      wait_async0();                       // chunk c+1 fully resident
      __syncthreads();                     // everyone done reading buf (c&1)
      if (c < N_CHUNK - 2)
        stage_chunk(w4f, c + 2, sW[c & 1], tid);   // overlap with chunk c+1 compute
    }
  }

  // ---- reduce lane pairs (v_lsb 0/1 partials) and store 56 outputs/edge ----
  float o[56];
#pragma unroll
  for (int w = 0; w < 8; ++w) {
    o[w] = acc0[w];
    o[8  + 3*w + 0] = a1x[w]; o[8  + 3*w + 1] = a1y[w]; o[8  + 3*w + 2] = a1z[w];
    o[32 + 3*w + 0] = a2x[w]; o[32 + 3*w + 1] = a2y[w]; o[32 + 3*w + 2] = a2z[w];
  }
#pragma unroll
  for (int k = 0; k < 56; ++k) o[k] += __shfl_xor(o[k], 16, 32);
  if (half == 0) {
    float* op = out + (long)e * 56;
#pragma unroll
    for (int q = 0; q < 14; ++q)
      *(float4*)(op + 4 * q) = make_float4(o[4*q], o[4*q+1], o[4*q+2], o[4*q+3]);
  }
}

extern "C" void kernel_launch(void* const* d_in, const int* in_sizes, int n_in,
                              void* d_out, int out_size, void* d_ws, size_t ws_size,
                              hipStream_t stream) {
  const float* nodes = (const float*)d_in[0];
  const float* efeat = (const float*)d_in[1];
  const float* w1    = (const float*)d_in[2];
  const float* w2    = (const float*)d_in[3];
  const float* w3    = (const float*)d_in[4];
  const float* w4    = (const float*)d_in[5];
  const int*   eidx  = (const int*)d_in[6];
  _Float16* w4f = (_Float16*)d_ws;
  _Float16* wbf = w4f + W4F_ELEMS;
  float* outp = (float*)d_out;

  prep_w4_frags<<<W4F_ELEMS / 256, 256, 0, stream>>>(w4, w4f);
  prep_w123_frags<<<WBF_ELEMS / 256, 256, 0, stream>>>(w1, w2, w3, wbf);
  fctp_fused<<<N_EDGES / 64, 128, 0, stream>>>(nodes, efeat, eidx, w4f, wbf, outp);
}
